// Mamba2Sequence_71296457113750
// MI455X (gfx1250) — compile-verified, matrix operand order checked by reading, not run on hardware
//
#include <hip/hip_runtime.h>
#include <hip/hip_bf16.h>

// ---------------- constants (from reference) ----------------
#define BATCH     4
#define SEQLEN    2048
#define D_MODEL   1024
#define D_STATE   64
#define D_CONV    4
#define HEADDIM   64
#define D_INNER   2048
#define NHEADS    32
#define CONV_DIM  2176        // D_INNER + 2*D_STATE
#define D_IN_PROJ 4256        // 2*D_INNER + 2*D_STATE + NHEADS
#define MROWS     (BATCH * SEQLEN)   // 8192

typedef __attribute__((ext_vector_type(16))) _Float16 v16h;
typedef __attribute__((ext_vector_type(8)))  _Float16 v8h;
typedef __attribute__((ext_vector_type(8)))  float    v8f;

#if defined(__AMDGCN__) && __has_builtin(__builtin_amdgcn_global_load_async_to_lds_b128) && \
    __has_builtin(__builtin_amdgcn_s_wait_asynccnt)
#define HAVE_ASYNC_LDS 1
typedef int v4i __attribute__((ext_vector_type(4)));
typedef __attribute__((address_space(1))) v4i as1_v4i;   // global
typedef __attribute__((address_space(3))) v4i as3_v4i;   // LDS
#else
#define HAVE_ASYNC_LDS 0
#endif

// ---------------- f32 -> f16 convert ----------------
__global__ void cvt_f32_f16(const float* __restrict__ in, _Float16* __restrict__ out, size_t n) {
    size_t i = (size_t)blockIdx.x * blockDim.x + threadIdx.x;
    if (i < n) out[i] = (_Float16)in[i];
}

// in[K][N] (row-major f32)  ->  out[N][K] (row-major f16)
__global__ void cvt_transpose_f16(const float* __restrict__ in, _Float16* __restrict__ out,
                                  int K, int N) {
    size_t i = (size_t)blockIdx.x * blockDim.x + threadIdx.x;
    size_t total = (size_t)K * N;
    if (i < total) {
        int k = (int)(i / N);
        int n = (int)(i % N);
        out[(size_t)n * K + k] = (_Float16)in[i];
    }
}

// ---------------- WMMA f16 GEMM:  C[M][N] = A[M][K] * Bt[N][K]^T ----------------
// 32x32 register block per wave32: 2 A-frags x 2 B-frags -> 4 independent wmmas
// per 8 b128 loads (2x arithmetic intensity vs one-tile-per-wave).

// Build a 16-bit A/B fragment per ISA 7.12.2:
//  lanes 0-15:  row = lane,    K {0..7} in v0..3, K {16..23} in v4..7
//  lanes 16-31: row = lane-16, K {8..15} and {24..31}
static __device__ __forceinline__ v16h load_frag(const _Float16* p) {
    v8h lo = *(const v8h*)(p);        // K koff..koff+7
    v8h hi = *(const v8h*)(p + 16);   // K koff+16..koff+23
    v16h f;
#pragma unroll
    for (int i = 0; i < 8; ++i) { f[i] = lo[i]; f[i + 8] = hi[i]; }
    return f;
}

static __device__ __forceinline__ void store_tile(float* C, int N, int tm, int tn,
                                                  int lane, const v8f& c) {
    const int n     = tn + (lane & 15);
    const int mbase = tm + ((lane >> 4) << 3);   // C/D layout: lane>=16 -> M = 8+r
#pragma unroll
    for (int r = 0; r < 8; ++r)
        C[(size_t)(mbase + r) * N + n] = c[r];
}

__global__ void gemm_f16_wmma(const _Float16* __restrict__ A,
                              const _Float16* __restrict__ Bt,
                              float* __restrict__ C,
                              int M, int N, int K) {
    const int wave  = threadIdx.x >> 5;
    const int lane  = threadIdx.x & 31;
    const int tileM = blockIdx.x * 32;
    const int tileN = blockIdx.y * 256 + wave * 32;
    if (tileN >= N) return;                      // uniform per wave -> EXEC stays all-1s

    const int r16  = lane & 15;
    const int koff = (lane >> 4) << 3;           // 0 or 8

    const _Float16* ap0 = A  + (size_t)(tileM + r16) * K + koff;
    const _Float16* ap1 = ap0 + (size_t)16 * K;
    const _Float16* bp0 = Bt + (size_t)(tileN + r16) * K + koff;
    const _Float16* bp1 = bp0 + (size_t)16 * K;

    v8f c00 = {}, c01 = {}, c10 = {}, c11 = {};
    for (int k = 0; k < K; k += 32) {
        v16h a0 = load_frag(ap0 + k);
        v16h a1 = load_frag(ap1 + k);
        v16h b0 = load_frag(bp0 + k);
        v16h b1 = load_frag(bp1 + k);
        c00 = __builtin_amdgcn_wmma_f32_16x16x32_f16(false, a0, false, b0, (short)0, c00, false, false);
        c01 = __builtin_amdgcn_wmma_f32_16x16x32_f16(false, a0, false, b1, (short)0, c01, false, false);
        c10 = __builtin_amdgcn_wmma_f32_16x16x32_f16(false, a1, false, b0, (short)0, c10, false, false);
        c11 = __builtin_amdgcn_wmma_f32_16x16x32_f16(false, a1, false, b1, (short)0, c11, false, false);
    }

    store_tile(C, N, tileM,      tileN,      lane, c00);
    store_tile(C, N, tileM,      tileN + 16, lane, c01);
    store_tile(C, N, tileM + 16, tileN,      lane, c10);
    store_tile(C, N, tileM + 16, tileN + 16, lane, c11);
}

// ---------------- dt = softplus(raw_dt + dt_bias) ----------------
__global__ void dt_softplus(const float* __restrict__ zxbcdt,
                            const float* __restrict__ dt_bias,
                            float* __restrict__ dtbuf) {
    int i = blockIdx.x * blockDim.x + threadIdx.x;   // MROWS*NHEADS
    if (i < MROWS * NHEADS) {
        int row = i >> 5, h = i & 31;
        float v = zxbcdt[(size_t)row * D_IN_PROJ + (2 * D_INNER + 2 * D_STATE) + h] + dt_bias[h];
        dtbuf[i] = (v > 20.f) ? v : log1pf(__expf(v));
    }
}

// ---------------- depthwise causal conv(d_conv=4) + silu ----------------
__global__ void conv_silu(const float* __restrict__ zxbcdt,
                          const float* __restrict__ conv_w,
                          const float* __restrict__ conv_b,
                          float* __restrict__ xBC) {
    int c = blockIdx.y * blockDim.x + threadIdx.x;
    if (c >= CONV_DIM) return;
    int bl = blockIdx.x;                 // b*SEQLEN + l
    int b = bl / SEQLEN, l = bl % SEQLEN;
    float acc = conv_b[c];
#pragma unroll
    for (int k = 0; k < D_CONV; ++k) {
        int lt = l - (D_CONV - 1) + k;
        if (lt >= 0)
            acc += zxbcdt[((size_t)b * SEQLEN + lt) * D_IN_PROJ + D_INNER + c] * conv_w[c * D_CONV + k];
    }
    xBC[(size_t)bl * CONV_DIM + c] = acc / (1.f + __expf(-acc));
}

// ---------------- sequential SSM scan: one block per (batch, head) ----------------
// thread (p = tid/4, 16 n-values at (tid&3)*16); state in registers;
// B/C/x staged into LDS via CDNA5 async-to-LDS (ASYNCcnt) when available.
__global__ void ssm_scan(const float* __restrict__ xBC,
                         const float* __restrict__ dtbuf,
                         const float* __restrict__ A_log,
                         const float* __restrict__ Dp,
                         float* __restrict__ y) {
    const int b   = blockIdx.x >> 5;     // / NHEADS
    const int h   = blockIdx.x & 31;
    const int tid = threadIdx.x;         // 256
    const int p   = tid >> 2;
    const int ng  = tid & 3;
    const int n0  = ng << 4;

    const float Ah = -__expf(A_log[h]);
    const float Dh = Dp[h];

    float st[16];
#pragma unroll
    for (int j = 0; j < 16; ++j) st[j] = 0.f;

    // stage: [0..63]=B, [64..127]=C, [128..191]=x   (48 lanes x 16B)
    __shared__ __align__(16) float stage[192];
    __shared__ float sdt;
    float* Bs = stage;
    float* Cs = stage + 64;
    float* Xs = stage + 128;

    const float* xbase = xBC + (size_t)b * SEQLEN * CONV_DIM;

    for (int t = 0; t < SEQLEN; ++t) {
        const float* row = xbase + (size_t)t * CONV_DIM;
#if HAVE_ASYNC_LDS
        if (tid < 48) {
            const float* g;
            if (tid < 16)      g = row + D_INNER + tid * 4;
            else if (tid < 32) g = row + D_INNER + D_STATE + (tid - 16) * 4;
            else               g = row + h * HEADDIM + (tid - 32) * 4;
            __builtin_amdgcn_global_load_async_to_lds_b128(
                (as1_v4i*)g, (as3_v4i*)&stage[tid * 4], 0, 0);
        }
        if (tid == 192) sdt = dtbuf[((size_t)b * SEQLEN + t) * NHEADS + h];
        __builtin_amdgcn_s_wait_asynccnt(0);
        __syncthreads();
#else
        if (tid < 64)        Bs[tid]        = row[D_INNER + tid];
        else if (tid < 128)  Cs[tid - 64]   = row[D_INNER + D_STATE + (tid - 64)];
        else if (tid < 192)  Xs[tid - 128]  = row[h * HEADDIM + (tid - 128)];
        else if (tid == 192) sdt            = dtbuf[((size_t)b * SEQLEN + t) * NHEADS + h];
        __syncthreads();
#endif
        // prefetch next timestep's row while this step computes (latency-bound chain)
        if (t + 1 < SEQLEN) {
            const float* nrow = row + CONV_DIM;
            if (tid == 224)      __builtin_prefetch(nrow + D_INNER, 0, 0);
            else if (tid == 228) __builtin_prefetch(nrow + h * HEADDIM, 0, 0);
        }

        const float dtv  = sdt;
        const float dA   = __expf(dtv * Ah);
        const float xp   = Xs[p];
        const float coef = dtv * xp;
        float acc = 0.f;
#pragma unroll
        for (int j = 0; j < 16; ++j) {
            st[j] = st[j] * dA + coef * Bs[n0 + j];
            acc  += st[j] * Cs[n0 + j];
        }
        acc += __shfl_xor(acc, 1, 32);
        acc += __shfl_xor(acc, 2, 32);
        if (ng == 0)
            y[((size_t)b * SEQLEN + t) * D_INNER + h * HEADDIM + p] = acc + Dh * xp;
        __syncthreads();
    }
}

// ---------------- y * silu(z), RMSNorm, scale by norm_w, emit f16 ----------------
__global__ void gate_rmsnorm(const float* __restrict__ y,
                             const float* __restrict__ zxbcdt,
                             const float* __restrict__ norm_w,
                             _Float16* __restrict__ out) {
    const int row = blockIdx.x;          // MROWS
    const int tid = threadIdx.x;         // 256, 8 elems each
    const float* yr = y      + (size_t)row * D_INNER;
    const float* zr = zxbcdt + (size_t)row * D_IN_PROJ;

    float vals[8];
    float ss = 0.f;
#pragma unroll
    for (int i = 0; i < 8; ++i) {
        int c = tid + i * 256;
        float z = zr[c];
        float v = yr[c] * (z / (1.f + __expf(-z)));
        vals[i] = v;
        ss += v * v;
    }
    __shared__ float red[256];
    red[tid] = ss;
    __syncthreads();
    for (int s = 128; s > 0; s >>= 1) {
        if (tid < s) red[tid] += red[tid + s];
        __syncthreads();
    }
    const float scale = rsqrtf(red[0] / (float)D_INNER + 1e-5f);
#pragma unroll
    for (int i = 0; i < 8; ++i) {
        int c = tid + i * 256;
        out[(size_t)row * D_INNER + c] = (_Float16)(vals[i] * scale * norm_w[c]);
    }
}

// ---------------- launch ----------------
extern "C" void kernel_launch(void* const* d_in, const int* in_sizes, int n_in,
                              void* d_out, int out_size, void* d_ws, size_t ws_size,
                              hipStream_t stream) {
    const float* x       = (const float*)d_in[0];
    const float* W_in    = (const float*)d_in[1];
    const float* conv_w  = (const float*)d_in[2];
    const float* conv_b  = (const float*)d_in[3];
    const float* dt_bias = (const float*)d_in[4];
    const float* A_log   = (const float*)d_in[5];
    const float* Dvec    = (const float*)d_in[6];
    const float* norm_w  = (const float*)d_in[7];
    const float* W_out   = (const float*)d_in[8];
    float* out = (float*)d_out;

    // workspace layout (bytes)
    char* ws = (char*)d_ws;
    float*    zxbcdt = (float*)ws;                              // 8192*4256*4 = 139,460,608
    float*    xBC    = (float*)(ws + 139460608ull);             // 8192*2176*4 =  71,303,168
    float*    dtbuf  = (float*)(ws + 210763776ull);             // 8192*32*4   =   1,048,576
    float*    ybuf   = (float*)(ws + 211812352ull);             // 8192*2048*4 =  67,108,864
    _Float16* hA     = (_Float16*)(ws + 278921216ull);          // max 33,554,432 (x_f16 / ynorm_f16)
    _Float16* hB     = (_Float16*)(ws + 312475648ull);          // max  8,716,288 (W_in^T / W_out^T f16)

    const int TB = 256;

    // 1) x -> f16
    {
        size_t n = (size_t)MROWS * D_MODEL;
        cvt_f32_f16<<<(unsigned)((n + TB - 1) / TB), TB, 0, stream>>>(x, hA, n);
    }
    // 2) W_in (1024 x 4256) -> transposed f16 (4256 x 1024)
    {
        size_t n = (size_t)D_MODEL * D_IN_PROJ;
        cvt_transpose_f16<<<(unsigned)((n + TB - 1) / TB), TB, 0, stream>>>(W_in, hB, D_MODEL, D_IN_PROJ);
    }
    // 3) zxbcdt = x @ W_in    (M=8192, N=4256, K=1024); 32x32 tile/wave, 256-wide block tile
    {
        dim3 grid(MROWS / 32, (D_IN_PROJ + 255) / 256);
        gemm_f16_wmma<<<grid, TB, 0, stream>>>(hA, hB, zxbcdt, MROWS, D_IN_PROJ, D_MODEL);
    }
    // 4) dt softplus
    dt_softplus<<<(MROWS * NHEADS + TB - 1) / TB, TB, 0, stream>>>(zxbcdt, dt_bias, dtbuf);
    // 5) conv + silu
    {
        dim3 grid(MROWS, (CONV_DIM + TB - 1) / TB);
        conv_silu<<<grid, TB, 0, stream>>>(zxbcdt, conv_w, conv_b, xBC);
    }
    // 6) sequential scan
    ssm_scan<<<BATCH * NHEADS, TB, 0, stream>>>(xBC, dtbuf, A_log, Dvec, ybuf);
    // 7) gate + RMSNorm -> f16
    gate_rmsnorm<<<MROWS, TB, 0, stream>>>(ybuf, zxbcdt, norm_w, hA);
    // 8) W_out (2048 x 1024) -> transposed f16 (1024 x 2048)
    {
        size_t n = (size_t)D_INNER * D_MODEL;
        cvt_transpose_f16<<<(unsigned)((n + TB - 1) / TB), TB, 0, stream>>>(W_out, hB, D_INNER, D_MODEL);
    }
    // 9) out = y_norm @ W_out  (M=8192, N=1024, K=2048)
    {
        dim3 grid(MROWS / 32, (D_MODEL + 255) / 256);
        gemm_f16_wmma<<<grid, TB, 0, stream>>>(hA, hB, out, MROWS, D_MODEL, D_INNER);
    }
}
